// MultiLayerGraphSAGE_48773648613819
// MI455X (gfx1250) — matrix operand order repaired and checked by reading, not compiled, not run on hardware
//
#include <hip/hip_runtime.h>

// ---------------------------------------------------------------------------
// MultiLayerGraphSAGE for gfx1250 (MI455X, wave32, WMMA)
//   h = relu(mean_agg(h)[dst] @ Wl + bias + h @ Wr), 3 layers
// GEMM path: fused A=[mean|x] (f16) x Bt=[Wl;Wr]^T (f16), f32 accum via
// v_wmma_f32_16x16x32_f16. Scatter path: L2-resident f32 atomics.
// This revision decouples B-fragment loads from WMMA issue so each k-step
// has one load clause + one wait instead of a wait per WMMA.
// ---------------------------------------------------------------------------

#define N_NODES 50000
#define N_EDGES 800000

typedef _Float16 half8_t __attribute__((ext_vector_type(8)));
typedef _Float16 v16h    __attribute__((ext_vector_type(16)));
typedef float    v8f     __attribute__((ext_vector_type(8)));

// ---------------- zero fill ----------------
__global__ void zero_f32(float* __restrict__ p, int n) {
    int i = blockIdx.x * blockDim.x + threadIdx.x;
    if (i < n) p[i] = 0.0f;
}

// ---------------- degree count ----------------
__global__ void count_deg(const int* __restrict__ dst, float* __restrict__ cnt, int E) {
    int i = blockIdx.x * blockDim.x + threadIdx.x;
    if (i < E) atomicAdd(&cnt[dst[i]], 1.0f);
}

// ---------------- scatter-add of source features ----------------
// thread = (edge, 4-float chunk). Gather float4 from h[src], atomic-add to agg[dst].
__global__ void scatter4(const float* __restrict__ h, const int* __restrict__ src,
                         const int* __restrict__ dst, float* __restrict__ agg,
                         int E, int F, int logC) {
    int idx = blockIdx.x * blockDim.x + threadIdx.x;
    int total = E << logC;
    if (idx >= total) return;
    int e = idx >> logC;
    int c = (idx & ((1 << logC) - 1)) << 2;     // float offset within feature row
    int s = src[e];
    int d = dst[e];
    float4 v = *reinterpret_cast<const float4*>(h + (size_t)s * F + c);
    float* o = agg + (size_t)d * F + c;
    atomicAdd(o + 0, v.x);
    atomicAdd(o + 1, v.y);
    atomicAdd(o + 2, v.z);
    atomicAdd(o + 3, v.w);
}

// ---------------- mean + pack A = [mean | x] as f16 ----------------
__global__ void finalizeA(const float* __restrict__ agg, const float* __restrict__ cnt,
                          const float* __restrict__ h, _Float16* __restrict__ Ah,
                          int F, int logF, int total) {
    int idx = blockIdx.x * blockDim.x + threadIdx.x;
    if (idx >= total) return;
    int n = idx >> logF;
    int f = idx & (F - 1);
    float c = cnt[n];
    float inv = 1.0f / fmaxf(c, 1.0f);
    size_t base = (size_t)n * (2 * F);
    Ah[base + f]     = (_Float16)(agg[(size_t)n * F + f] * inv);
    Ah[base + F + f] = (_Float16)(h[(size_t)n * F + f]);
}

// ---------------- pack Bt[o][k] : k<Fi -> Wl[k][o], else Wr[k-Fi][o] ----------------
__global__ void convertBt(const float* __restrict__ Wl, const float* __restrict__ Wr,
                          _Float16* __restrict__ Bt, int Fi, int Fo, int logFo, int total) {
    int idx = blockIdx.x * blockDim.x + threadIdx.x;
    if (idx >= total) return;
    int k = idx >> logFo;
    int o = idx & (Fo - 1);
    int K2 = 2 * Fi;
    Bt[(size_t)o * K2 + k]      = (_Float16)Wl[(size_t)k * Fo + o];
    Bt[(size_t)o * K2 + Fi + k] = (_Float16)Wr[(size_t)k * Fo + o];
}

// ---------------- fragment loaders ----------------
__device__ __forceinline__ v16h load_frag16(const _Float16* p0, const _Float16* p1) {
    half8_t lo = *reinterpret_cast<const half8_t*>(p0);
    half8_t hi = *reinterpret_cast<const half8_t*>(p1);
    return __builtin_shufflevector(lo, hi,
                                   0, 1, 2, 3, 4, 5, 6, 7,
                                   8, 9, 10, 11, 12, 13, 14, 15);
}

// ---------------- WMMA GEMM: out = relu(A @ Bt^T + bias) ----------------
// A: M x K2 (f16, row-major), Bt: N x K2 (f16, row-major; Bt = B^T)
// One wave32 computes a 16x64 output slab (4 WMMA accumulators sharing 1 A frag).
// M = 3125*16 exactly; N % 64 == 0; K2 % 32 == 0 -> no tails, EXEC all-ones.
__global__ void __launch_bounds__(256)
sage_gemm_wmma(const _Float16* __restrict__ A, const _Float16* __restrict__ Bt,
               const float* __restrict__ bias, float* __restrict__ out,
               int M, int N, int K2, int totalWaves) {
    const int lane = threadIdx.x & 31;
    const int wave = blockIdx.x * (blockDim.x >> 5) + (threadIdx.x >> 5);
    if (wave >= totalWaves) return;

    const int ngroups = N >> 6;               // number of 64-col groups
    const int mt = wave / ngroups;            // 16-row tile index
    const int ng = wave % ngroups;            // 64-col group index

    const int lo16 = lane & 15;
    const int hsel = lane >> 4;               // 0: lanes 0-15, 1: lanes 16-31
    const int row  = mt * 16 + lo16;          // A row this lane feeds
    const int akb  = hsel * 16;               // A K sub-offset (contiguous, matches B)
    const int bkb  = hsel * 16;               // B K sub-offset (contiguous, matches A)

    v8f zero8 = {0.f, 0.f, 0.f, 0.f, 0.f, 0.f, 0.f, 0.f};
    v8f acc0 = zero8, acc1 = zero8, acc2 = zero8, acc3 = zero8;

    const _Float16* arow = A + (size_t)row * K2;
    const _Float16* brow0 = Bt + (size_t)(ng * 64 +  0 + lo16) * K2 + bkb;
    const _Float16* brow1 = Bt + (size_t)(ng * 64 + 16 + lo16) * K2 + bkb;
    const _Float16* brow2 = Bt + (size_t)(ng * 64 + 32 + lo16) * K2 + bkb;
    const _Float16* brow3 = Bt + (size_t)(ng * 64 + 48 + lo16) * K2 + bkb;

    for (int kk = 0; kk < K2; kk += 32) {
        // Issue ALL fragment loads first (distinct registers) -> one load
        // clause + one wait per k-step instead of a wait per WMMA.
        const _Float16* ap = arow + kk + akb;
        v16h a  = load_frag16(ap, ap + 8);
        v16h b0 = load_frag16(brow0 + kk, brow0 + kk + 8);
        v16h b1 = load_frag16(brow1 + kk, brow1 + kk + 8);
        v16h b2 = load_frag16(brow2 + kk, brow2 + kk + 8);
        v16h b3 = load_frag16(brow3 + kk, brow3 + kk + 8);

        acc0 = __builtin_amdgcn_wmma_f32_16x16x32_f16(
            false, a, false, b0, (short)0, acc0, false, false);
        acc1 = __builtin_amdgcn_wmma_f32_16x16x32_f16(
            false, a, false, b1, (short)0, acc1, false, false);
        acc2 = __builtin_amdgcn_wmma_f32_16x16x32_f16(
            false, a, false, b2, (short)0, acc2, false, false);
        acc3 = __builtin_amdgcn_wmma_f32_16x16x32_f16(
            false, a, false, b3, (short)0, acc3, false, false);
    }

    // D layout: VGPR r, lane L -> D[M = r + 8*(L>>4)][N = L&15]
    const int rbase = mt * 16 + 8 * hsel;
    v8f accs[4] = {acc0, acc1, acc2, acc3};
#pragma unroll
    for (int j = 0; j < 4; ++j) {
        int col = ng * 64 + j * 16 + lo16;
        float bv = bias[col];
#pragma unroll
        for (int r = 0; r < 8; ++r) {
            float v = accs[j][r] + bv;
            out[(size_t)(rbase + r) * N + col] = v > 0.0f ? v : 0.0f;
        }
    }
}

// ---------------------------------------------------------------------------
// Host-side layer driver
// ---------------------------------------------------------------------------
static void run_layer(const float* hin, const int* eall, const float* Wl,
                      const float* bl, const float* Wr, float* hout,
                      int Fi, int Fo, float* agg, float* cnt,
                      _Float16* Ah, _Float16* Bt, hipStream_t stream) {
    const int E = N_EDGES;
    const int* src = eall;          // edge[0] (flow: aggregate edge[0] -> edge[1])
    const int* dst = eall + E;      // edge[1]
    const int K2 = 2 * Fi;

    int nAgg = N_NODES * Fi;
    zero_f32<<<(nAgg + 255) / 256, 256, 0, stream>>>(agg, nAgg);
    zero_f32<<<(N_NODES + 255) / 256, 256, 0, stream>>>(cnt, N_NODES);

    count_deg<<<(E + 255) / 256, 256, 0, stream>>>(dst, cnt, E);

    int logC = (Fi == 128) ? 5 : 6;             // log2(Fi/4)
    int totS = E << logC;
    scatter4<<<(totS + 255) / 256, 256, 0, stream>>>(hin, src, dst, agg, E, Fi, logC);

    int logF = (Fi == 128) ? 7 : 8;
    int nNF = N_NODES * Fi;
    finalizeA<<<(nNF + 255) / 256, 256, 0, stream>>>(agg, cnt, hin, Ah, Fi, logF, nNF);

    int logFo = (Fo == 256) ? 8 : 7;
    int nB = Fi * Fo;
    convertBt<<<(nB + 255) / 256, 256, 0, stream>>>(Wl, Wr, Bt, Fi, Fo, logFo, nB);

    int waves = (N_NODES / 16) * (Fo / 64);     // 16x64 slab per wave
    int blocks = (waves + 7) / 8;               // 8 waves (256 threads) per block
    sage_gemm_wmma<<<blocks, 256, 0, stream>>>(Ah, Bt, bl, hout, N_NODES, Fo, K2, waves);
}

extern "C" void kernel_launch(void* const* d_in, const int* in_sizes, int n_in,
                              void* d_out, int out_size, void* d_ws, size_t ws_size,
                              hipStream_t stream) {
    (void)in_sizes; (void)n_in; (void)out_size; (void)ws_size;

    const float* x     = (const float*)d_in[0];
    const int*   edge0 = (const int*)d_in[1];
    const int*   edge1 = (const int*)d_in[2];
    const int*   edge2 = (const int*)d_in[3];
    const float* Wl0 = (const float*)d_in[4];
    const float* bl0 = (const float*)d_in[5];
    const float* Wr0 = (const float*)d_in[6];
    const float* Wl1 = (const float*)d_in[7];
    const float* bl1 = (const float*)d_in[8];
    const float* Wr1 = (const float*)d_in[9];
    const float* Wl2 = (const float*)d_in[10];
    const float* bl2 = (const float*)d_in[11];
    const float* Wr2 = (const float*)d_in[12];
    float* out = (float*)d_out;

    // Workspace layout (bytes, 16B-aligned offsets), total ~154 MB:
    char* ws = (char*)d_ws;
    float*    agg  = (float*)(ws);                     // N * 256 f32 = 51.2 MB
    _Float16* Ah   = (_Float16*)(ws + 51200000);       // N * 512 f16 = 51.2 MB
    float*    htmp = (float*)(ws + 102400000);         // N * 256 f32 = 51.2 MB
    float*    cnt  = (float*)(ws + 153600000);         // N f32      = 0.2 MB
    _Float16* Bt   = (_Float16*)(ws + 153800192);      // 512*256 f16 = 0.25 MB

    // layer 0: 128 -> 256
    run_layer(x,    edge0, Wl0, bl0, Wr0, htmp, 128, 256, agg, cnt, Ah, Bt, stream);
    // layer 1: 256 -> 256 (htmp in/out is safe: GEMM writes after scatter/finalize read)
    run_layer(htmp, edge1, Wl1, bl1, Wr1, htmp, 256, 256, agg, cnt, Ah, Bt, stream);
    // layer 2: 256 -> 128
    run_layer(htmp, edge2, Wl2, bl2, Wr2, out,  256, 128, agg, cnt, Ah, Bt, stream);
}
